// DynamicTopoNet_10299331575868
// MI455X (gfx1250) — matrix operand order repaired and checked by reading, not compiled, not run on hardware
//
#include <hip/hip_runtime.h>
#include <math.h>

// ---------- types ----------
typedef __attribute__((ext_vector_type(16))) __bf16 v16bf;
typedef __attribute__((ext_vector_type(8)))  __bf16 v8bf;
typedef __attribute__((ext_vector_type(8)))  float  v8f;
typedef int b128_t __attribute__((vector_size(16)));  // matches async-LDS builtin pointee

#define DIM  512
#define NTOK 8192
#define SEQ  2048
#define NB   4
#define NH   8
#define HD   64

// ---------- bf16 helpers: native HW converts on gfx1250 ----------
__device__ __forceinline__ __bf16 f2bf(float f)  { return (__bf16)f; }
__device__ __forceinline__ float  bf2f(__bf16 b) { return (float)b; }
__device__ __forceinline__ __bf16 to_bf(float v)  { return (__bf16)v; }
__device__ __forceinline__ __bf16 to_bf(__bf16 v) { return v; }

// ---------- async memory -> LDS (CDNA5 GLOBAL_LOAD_ASYNC_TO_LDS_B128) ----------
#if __has_builtin(__builtin_amdgcn_global_load_async_to_lds_b128)
#define HAVE_ASYNC_LDS 1
#else
#define HAVE_ASYNC_LDS 0
#endif

// copy 8 bf16 (16 bytes) global -> LDS
__device__ __forceinline__ void lds_copy16(__bf16* dst, const __bf16* src) {
#if HAVE_ASYNC_LDS
  __builtin_amdgcn_global_load_async_to_lds_b128(
      (__attribute__((address_space(1))) b128_t*)(src),
      (__attribute__((address_space(3))) b128_t*)(dst), 0, 0);
#else
  *(v8bf*)dst = *(const v8bf*)src;
#endif
}
__device__ __forceinline__ void lds_copy_fence() {
#if HAVE_ASYNC_LDS
#if __has_builtin(__builtin_amdgcn_s_wait_asynccnt)
  __builtin_amdgcn_s_wait_asynccnt(0);
#else
  asm volatile("s_wait_asynccnt 0" ::: "memory");
#endif
#endif
}

// ---------- WMMA fragment loaders (row-major source tiles) ----------
// A 16x32 bf16: lane<16 holds row=lane, K chunks {0..7,16..23}; lane>=16: {8..15,24..31}
__device__ __forceinline__ v16bf load_a_frag(const __bf16* base, int ld) {
  int lane = threadIdx.x & 31;
  const __bf16* p = base + (lane & 15) * ld + ((lane >> 4) << 3);
  v8bf lo = *(const v8bf*)p;
  v8bf hi = *(const v8bf*)(p + 16);
  return __builtin_shufflevector(lo, hi, 0, 1, 2, 3, 4, 5, 6, 7,
                                 8, 9, 10, 11, 12, 13, 14, 15);
}
// B 32x16 bf16: lane n holds col n (= source row n), K chunks {0..7,16..23} (lane<16) / {8..15,24..31} (lane>=16)
__device__ __forceinline__ v16bf load_b_frag(const __bf16* base, int ld) {
  int lane = threadIdx.x & 31;
  const __bf16* p = base + (lane & 15) * ld + ((lane >> 4) << 3);
  v8bf lo = *(const v8bf*)p;
  v8bf hi = *(const v8bf*)(p + 16);
  return __builtin_shufflevector(lo, hi, 0, 1, 2, 3, 4, 5, 6, 7,
                                 8, 9, 10, 11, 12, 13, 14, 15);
}
__device__ __forceinline__ v8f wmma_bf16(v16bf a, v16bf b, v8f c) {
  return __builtin_amdgcn_wmma_f32_16x16x32_bf16(false, a, false, b, (short)0, c,
                                                 false, false);
}
__device__ __forceinline__ float redsum16(float v) {
  v += __shfl_xor(v, 1, 16); v += __shfl_xor(v, 2, 16);
  v += __shfl_xor(v, 4, 16); v += __shfl_xor(v, 8, 16);
  return v;
}
__device__ __forceinline__ float redmax16(float v) {
  v = fmaxf(v, __shfl_xor(v, 1, 16)); v = fmaxf(v, __shfl_xor(v, 2, 16));
  v = fmaxf(v, __shfl_xor(v, 4, 16)); v = fmaxf(v, __shfl_xor(v, 8, 16));
  return v;
}

// ---------- generic GEMM: C[n,e] = sum_k A[n,k] * W[e,k] (+bias) ----------
// Block tile 128x64, 8 waves; wave w owns rows [16w,16w+16), 4 accum tiles of 16 cols.
#define LDA 40
#define LDW 40
template <int MODE, typename AT>
__global__ __launch_bounds__(256) void gemm_xw(
    const AT* __restrict__ A, const float* __restrict__ W,
    const float* __restrict__ bias, int K,
    float* __restrict__ outf, __bf16* __restrict__ ob0, __bf16* __restrict__ ob1,
    __bf16* __restrict__ ob2, float* __restrict__ sq) {
  __shared__ __bf16 At[128 * LDA];
  __shared__ __bf16 Wt[64 * LDW];
  const int tid = threadIdx.x, lane = tid & 31, wid = tid >> 5;
  const int rowbase = blockIdx.x * 128;
  const int colbase = blockIdx.y * 64;
  const int ar = tid >> 1, ac = (tid & 1) * 16;      // A-tile slice owned by thread
  const int wr = tid >> 2, wc = (tid & 3) * 8;       // W-tile slice owned by thread
  v8f acc[4] = {};
  for (int k0 = 0; k0 < K; k0 += 32) {
    __syncthreads();
    {  // A tile 128x32
      const AT* src = A + (size_t)(rowbase + ar) * K + k0 + ac;
      __bf16* dst = At + ar * LDA + ac;
      if constexpr (sizeof(AT) == 2) {
        lds_copy16(dst, (const __bf16*)src);
        lds_copy16(dst + 8, (const __bf16*)src + 8);
      } else {
#pragma unroll
        for (int i = 0; i < 16; ++i) dst[i] = to_bf(src[i]);
      }
      if (k0 + 32 < K) __builtin_prefetch(src + 32, 0, 1);
    }
    {  // W tile 64x32 (fp32 weights -> bf16)
      const float* src = W + (size_t)(colbase + wr) * K + k0 + wc;
      __bf16* dst = Wt + wr * LDW + wc;
#pragma unroll
      for (int i = 0; i < 8; ++i) dst[i] = f2bf(src[i]);
      if (k0 + 32 < K) __builtin_prefetch(src + 32, 0, 1);
    }
    lds_copy_fence();
    __syncthreads();
    v16bf a = load_a_frag(At + (wid * 16) * LDA, LDA);
#pragma unroll
    for (int g = 0; g < 4; ++g) {
      v16bf b = load_b_frag(Wt + (g * 16) * LDW, LDW);
      acc[g] = wmma_bf16(a, b, acc[g]);
    }
  }
  const int rhalf = (lane >> 4) << 3;
#pragma unroll
  for (int r = 0; r < 8; ++r) {
    int n = rowbase + wid * 16 + r + rhalf;
    float ssum = 0.f;
#pragma unroll
    for (int g = 0; g < 4; ++g) {
      int e = colbase + g * 16 + (lane & 15);
      float c = acc[g][r] + bias[e];
      if constexpr (MODE == 0) {  // QKV scatter: q,k [B,H,S,hd]; v transposed [B,H,hd,S]
        int b_ = n / SEQ, s_ = n % SEQ;
        int sel = e / DIM, h = (e % DIM) / HD, d = e % HD;
        if (sel == 0) {
          ob0[(((size_t)(b_ * NH + h)) * SEQ + s_) * HD + d] = f2bf(c);
        } else if (sel == 1) {
          ob1[(((size_t)(b_ * NH + h)) * SEQ + s_) * HD + d] = f2bf(c);
        } else {
          ob2[(((size_t)(b_ * NH + h)) * HD + d) * SEQ + s_] = f2bf(c);
        }
      } else if constexpr (MODE == 1) {  // out-proj: f (fp32 + bf16) + row norms
        outf[(size_t)n * DIM + e] = c;
        ob0[(size_t)n * DIM + e] = f2bf(c);
        ssum += c * c;
      } else if constexpr (MODE == 2) {  // W1 pre-LN
        outf[(size_t)n * (2 * DIM) + e] = c;
      } else {  // W2 -> gen_out
        outf[(size_t)n * DIM + e] = c;
      }
    }
    if constexpr (MODE == 1) {
      ssum = redsum16(ssum);
      if ((lane & 15) == 0) atomicAdd(&sq[n], ssum);
    }
  }
}

// ---------- flash attention: 128 q-rows/WG, one (b,h) per blockIdx.y ----------
#define ALDK 72
__global__ __launch_bounds__(256) void attn_kernel(
    const __bf16* __restrict__ q, const __bf16* __restrict__ k,
    const __bf16* __restrict__ vT, __bf16* __restrict__ ctx) {
  __shared__ __bf16 Kt[64 * ALDK];
  __shared__ __bf16 Vt[64 * ALDK];
  __shared__ __bf16 Pt[8 * 16 * ALDK];
  const int tid = threadIdx.x, lane = tid & 31, wid = tid >> 5;
  const int bh = blockIdx.y;
  const int q0 = blockIdx.x * 128 + wid * 16;
  const size_t qkbase = (size_t)bh * SEQ * HD;
  const size_t vbase  = (size_t)bh * HD * SEQ;

  v16bf qa[2];
#pragma unroll
  for (int kk = 0; kk < 2; ++kk)
    qa[kk] = load_a_frag(q + qkbase + (size_t)q0 * HD + kk * 32, HD);

  v8f oacc[4] = {};
  float m[8], l[8];
#pragma unroll
  for (int r = 0; r < 8; ++r) { m[r] = -INFINITY; l[r] = 0.f; }

  const int tr = tid >> 2, tc = (tid & 3) * 16;
  for (int kb = 0; kb < SEQ / 64; ++kb) {
    __syncthreads();
    {  // K tile: 64 keys x 64 dims (async -> LDS)
      const __bf16* src = k + qkbase + (size_t)(kb * 64 + tr) * HD + tc;
      __bf16* dst = Kt + tr * ALDK + tc;
      lds_copy16(dst, src);
      lds_copy16(dst + 8, src + 8);
    }
    {  // V^T tile: 64 dims x 64 keys (async -> LDS)
      const __bf16* src = vT + vbase + (size_t)tr * SEQ + kb * 64 + tc;
      __bf16* dst = Vt + tr * ALDK + tc;
      lds_copy16(dst, src);
      lds_copy16(dst + 8, src + 8);
    }
    lds_copy_fence();
    __syncthreads();
    v8f sc[4] = {};
#pragma unroll
    for (int kk = 0; kk < 2; ++kk) {
#pragma unroll
      for (int g = 0; g < 4; ++g) {
        v16bf b = load_b_frag(Kt + (g * 16) * ALDK + kk * 32, ALDK);
        sc[g] = wmma_bf16(qa[kk], b, sc[g]);
      }
    }
#pragma unroll
    for (int g = 0; g < 4; ++g) {
#pragma unroll
      for (int r = 0; r < 8; ++r) sc[g][r] *= 0.125f;  // 1/sqrt(64)
    }
    // online softmax per row (VGPR r <-> row, lanes%16 <-> cols)
#pragma unroll
    for (int r = 0; r < 8; ++r) {
      float mx = fmaxf(fmaxf(sc[0][r], sc[1][r]), fmaxf(sc[2][r], sc[3][r]));
      mx = redmax16(mx);
      float newm = fmaxf(m[r], mx);
      float corr = __expf(m[r] - newm);
      m[r] = newm;
      float ps = 0.f;
#pragma unroll
      for (int g = 0; g < 4; ++g) {
        float p = __expf(sc[g][r] - newm);
        sc[g][r] = p;
        ps += p;
      }
      ps = redsum16(ps);
      l[r] = l[r] * corr + ps;
#pragma unroll
      for (int g = 0; g < 4; ++g) oacc[g][r] *= corr;
    }
    // P (C-layout) -> per-wave LDS -> A-fragments; same-wave DS is in-order
    __bf16* Pw = Pt + wid * 16 * ALDK;
#pragma unroll
    for (int g = 0; g < 4; ++g) {
#pragma unroll
      for (int r = 0; r < 8; ++r)
        Pw[(r + ((lane >> 4) << 3)) * ALDK + g * 16 + (lane & 15)] = f2bf(sc[g][r]);
    }
#pragma unroll
    for (int kk = 0; kk < 2; ++kk) {
      v16bf a = load_a_frag(Pw + kk * 32, ALDK);
#pragma unroll
      for (int dg = 0; dg < 4; ++dg) {
        v16bf b = load_b_frag(Vt + (dg * 16) * ALDK + kk * 32, ALDK);
        oacc[dg] = wmma_bf16(a, b, oacc[dg]);
      }
    }
  }
  const int h = bh & (NH - 1);
#pragma unroll
  for (int r = 0; r < 8; ++r) {
    int row = q0 + r + ((lane >> 4) << 3);
    int n = (bh >> 3) * SEQ + row;
    float inv = 1.f / l[r];
#pragma unroll
    for (int dg = 0; dg < 4; ++dg) {
      int col = h * HD + dg * 16 + (lane & 15);
      ctx[(size_t)n * DIM + col] = f2bf(oacc[dg][r] * inv);
    }
  }
}

// ---------- entropy gating: two-pass streamed Gram via WMMA ----------
#define ELD 520
#define BLD 40
__global__ __launch_bounds__(128) void entropy_kernel(
    const __bf16* __restrict__ fbf, const float* __restrict__ f,
    const float* __restrict__ sq, const float* __restrict__ temperature,
    const float* __restrict__ target_entropy, float* __restrict__ ctrl_out,
    __bf16* __restrict__ ctrlb) {
  extern __shared__ char smem[];
  __bf16* Ap = (__bf16*)smem;                                 // 64 x ELD
  __bf16* Bt = (__bf16*)(smem + 64 * ELD * 2);                // 64 x BLD
  float* csm = (float*)(smem + 64 * ELD * 2 + 64 * BLD * 2);  // 64 controls
  const int tid = threadIdx.x, lane = tid & 31, wid = tid >> 5;
  const int rowbase = blockIdx.x * 64;
  {  // pin this WG's 64-row bf16 panel in LDS (reused for all 8192 columns, twice)
    int r = tid >> 1, half = tid & 1;
    const __bf16* src = fbf + (size_t)(rowbase + r) * DIM + half * 256;
    __bf16* dst = Ap + r * ELD + half * 256;
#pragma unroll
    for (int i = 0; i < 32; ++i) lds_copy16(dst + i * 8, src + i * 8);
  }
  lds_copy_fence();
  __syncthreads();
  const float temp = temperature[0];
  const float inv2t2 = 1.f / (2.f * temp * temp);
  const float invT = 1.f / temp;
  const float tgt = target_entropy[0];
  const int rhalf = (lane >> 4) << 3;
  float sqi[8];
#pragma unroll
  for (int r = 0; r < 8; ++r) sqi[r] = sq[rowbase + wid * 16 + r + rhalf];
  float rs[8] = {}, invrs[8] = {}, ent[8] = {};
  const int br = tid >> 1, bc = (tid & 1) * 16;
  for (int pass = 0; pass < 2; ++pass) {
    for (int jb = 0; jb < NTOK / 64; ++jb) {
      v8f acc[4] = {};
      for (int ks = 0; ks < DIM / 32; ++ks) {
        __syncthreads();
        {  // B tile 64x32 (columns j of the Gram), async -> LDS
          const __bf16* src = fbf + (size_t)(jb * 64 + br) * DIM + ks * 32 + bc;
          __bf16* dst = Bt + br * BLD + bc;
          lds_copy16(dst, src);
          lds_copy16(dst + 8, src + 8);
        }
        lds_copy_fence();
        __syncthreads();
        v16bf a = load_a_frag(Ap + (wid * 16) * ELD + ks * 32, ELD);
#pragma unroll
        for (int g = 0; g < 4; ++g) {
          v16bf b = load_b_frag(Bt + (g * 16) * BLD, BLD);
          acc[g] = wmma_bf16(a, b, acc[g]);
        }
      }
#pragma unroll
      for (int g = 0; g < 4; ++g) {
        int j = jb * 64 + g * 16 + (lane & 15);
        float sqj = sq[j];
#pragma unroll
        for (int r = 0; r < 8; ++r) {
          float d2 = fmaxf(sqi[r] + sqj - 2.f * acc[g][r], 0.f);
          float kern = __expf(-d2 * inv2t2);
          if (pass == 0) {
            rs[r] += kern;
          } else {
            float p = kern * invrs[r];
            ent[r] -= p * __logf(p + 1e-6f);
          }
        }
      }
    }
    if (pass == 0) {
#pragma unroll
      for (int r = 0; r < 8; ++r) invrs[r] = 1.f / redsum16(rs[r]);
    }
  }
#pragma unroll
  for (int r = 0; r < 8; ++r) {
    float e = redsum16(ent[r]);
    float ctl = 1.f / (1.f + __expf((e - tgt) * invT));  // sigmoid(-(e-tgt)/T)
    int lrow = wid * 16 + r + rhalf;
    if ((lane & 15) == 0) {
      ctrl_out[rowbase + lrow] = ctl;
      csm[lrow] = ctl;
    }
  }
  __syncthreads();
  {  // ctrl = f * control -> bf16 for the MLP GEMM
    int r = tid >> 1, half = tid & 1;
    float ctl = csm[r];
    const float* src = f + (size_t)(rowbase + r) * DIM + half * 256;
    __bf16* dst = ctrlb + (size_t)(rowbase + r) * DIM + half * 256;
    for (int i = 0; i < 256; ++i) dst[i] = f2bf(src[i] * ctl);
  }
}

// ---------- LayerNorm(1024) + exact GELU ----------
__global__ __launch_bounds__(256) void ln_gelu_kernel(
    const float* __restrict__ hpre, const float* __restrict__ g,
    const float* __restrict__ b, __bf16* __restrict__ hb) {
  __shared__ float red[256];
  const int row = blockIdx.x, tid = threadIdx.x;
  const float* hr = hpre + (size_t)row * 1024;
  float v[4];
  float s = 0.f;
#pragma unroll
  for (int i = 0; i < 4; ++i) { v[i] = hr[tid + i * 256]; s += v[i]; }
  red[tid] = s;
  __syncthreads();
  for (int off = 128; off > 0; off >>= 1) {
    if (tid < off) red[tid] += red[tid + off];
    __syncthreads();
  }
  float mu = red[0] * (1.f / 1024.f);
  __syncthreads();
  float s2 = 0.f;
#pragma unroll
  for (int i = 0; i < 4; ++i) { float d = v[i] - mu; s2 += d * d; }
  red[tid] = s2;
  __syncthreads();
  for (int off = 128; off > 0; off >>= 1) {
    if (tid < off) red[tid] += red[tid + off];
    __syncthreads();
  }
  float rstd = rsqrtf(red[0] * (1.f / 1024.f) + 1e-5f);
#pragma unroll
  for (int i = 0; i < 4; ++i) {
    int c = tid + i * 256;
    float y = (v[i] - mu) * rstd * g[c] + b[c];
    float ge = 0.5f * y * (1.f + erff(y * 0.70710678118f));
    hb[(size_t)row * 1024 + c] = f2bf(ge);
  }
}

// ---------- host ----------
extern "C" void kernel_launch(void* const* d_in, const int* in_sizes, int n_in,
                              void* d_out, int out_size, void* d_ws, size_t ws_size,
                              hipStream_t stream) {
  (void)in_sizes; (void)n_in; (void)out_size; (void)ws_size;
  const float* x    = (const float*)d_in[0];
  const float* wqkv = (const float*)d_in[1];
  const float* bqkv = (const float*)d_in[2];
  const float* wo   = (const float*)d_in[3];
  const float* bo   = (const float*)d_in[4];
  const float* temp = (const float*)d_in[5];
  const float* tgt  = (const float*)d_in[6];
  const float* W1   = (const float*)d_in[7];
  const float* b1   = (const float*)d_in[8];
  const float* lng  = (const float*)d_in[9];
  const float* lnb  = (const float*)d_in[10];
  const float* W2   = (const float*)d_in[11];
  const float* b2   = (const float*)d_in[12];
  float* out = (float*)d_out;

  char* ws = (char*)d_ws;
  size_t off = 0;
  auto alloc = [&](size_t bytes) {
    void* p = ws + off;
    off = (off + bytes + 255) & ~(size_t)255;
    return p;
  };
  __bf16* qb    = (__bf16*)alloc((size_t)NTOK * DIM * 2);
  __bf16* kb    = (__bf16*)alloc((size_t)NTOK * DIM * 2);
  __bf16* vT    = (__bf16*)alloc((size_t)NTOK * DIM * 2);
  __bf16* ctx   = (__bf16*)alloc((size_t)NTOK * DIM * 2);
  float*  fbuf  = (float*)alloc((size_t)NTOK * DIM * 4);
  __bf16* fbf   = (__bf16*)alloc((size_t)NTOK * DIM * 2);
  float*  sq    = (float*)alloc((size_t)NTOK * 4);
  __bf16* ctrlb = (__bf16*)alloc((size_t)NTOK * DIM * 2);
  float*  hpre  = (float*)alloc((size_t)NTOK * 2 * DIM * 4);
  __bf16* hb    = (__bf16*)alloc((size_t)NTOK * 2 * DIM * 2);

  (void)hipMemsetAsync(sq, 0, (size_t)NTOK * 4, stream);

  // 1) QKV projection + scatter (V transposed)
  gemm_xw<0, float><<<dim3(NTOK / 128, (3 * DIM) / 64), 256, 0, stream>>>(
      x, wqkv, bqkv, DIM, nullptr, qb, kb, vT, nullptr);
  // 2) flash attention
  attn_kernel<<<dim3(SEQ / 128, NB * NH), 256, 0, stream>>>(qb, kb, vT, ctx);
  // 3) out projection -> f (fp32+bf16) + row norms
  gemm_xw<1, __bf16><<<dim3(NTOK / 128, DIM / 64), 256, 0, stream>>>(
      ctx, wo, bo, DIM, fbuf, fbf, nullptr, nullptr, sq);
  // 4) entropy gating (Gram via WMMA, two passes) -> control + ctrl_bf16
  size_t esm = (size_t)64 * ELD * 2 + (size_t)64 * BLD * 2 + 64 * 4;
  entropy_kernel<<<NTOK / 64, 128, esm, stream>>>(
      fbf, fbuf, sq, temp, tgt, out + (size_t)NTOK * DIM, ctrlb);
  // 5) MLP up
  gemm_xw<2, __bf16><<<dim3(NTOK / 128, (2 * DIM) / 64), 256, 0, stream>>>(
      ctrlb, W1, b1, DIM, hpre, nullptr, nullptr, nullptr, nullptr);
  // 6) LayerNorm + GELU
  ln_gelu_kernel<<<NTOK, 256, 0, stream>>>(hpre, lng, lnb, hb);
  // 7) MLP down -> gen_out
  gemm_xw<3, __bf16><<<dim3(NTOK / 128, DIM / 64), 256, 0, stream>>>(
      hb, W2, b2, 2 * DIM, out, nullptr, nullptr, nullptr, nullptr);
}
